// Model_8452495638921
// MI455X (gfx1250) — compile-verified, matrix operand order checked by reading, not run on hardware
//
#include <hip/hip_runtime.h>
#include <hip/hip_bf16.h>
#include <cstdint>
#include <cstddef>

// ---- problem constants ----
#define L_   256
#define D_   1024
#define OD_  1024
#define NS_  10
#define ND_  10
#define SL_  5
#define NC_  7
#define NR_  16
#define NB_  8
#define E_   2048
#define SW_  5
#define WW_  11
#define H_   8
#define DH_  512

typedef __attribute__((ext_vector_type(16))) _Float16 v16h;
typedef __attribute__((ext_vector_type(8)))  float    v8f;

#define NEG_INF (-__builtin_inff())

// =====================================================================
// Core tiled WMMA GEMM:  C[M,N] = act( (accum?C:0) + rowscale[m]*(A@B) + bias[n] )
// A: MxK f16 row-major, B: KxN f16 row-major, C: MxN f32.
// Requires M%32==0, N%64==0, K%32==0 (all call sites satisfy this).
// Block = 256 threads = 8 waves; block tile 32(M) x 64(N); BK=32.
// Double-buffered pipeline: A tile staged memory->LDS via
// GLOBAL_LOAD_ASYNC_TO_LDS_B64 (ASYNCcnt) overlapping the WMMA of the
// previous tile; B tile staged via VGPRs with a 16-bit transpose into LDS.
// Row strides padded to 40 halfs so A and B fragments load as ds_load_b128.
// Each wave computes one 16x16 tile via v_wmma_f32_16x16x32_f16.
// =====================================================================
__global__ __launch_bounds__(256) void k_gemm(
    const _Float16* __restrict__ A, const _Float16* __restrict__ B,
    float* __restrict__ C, const float* __restrict__ bias,
    const float* __restrict__ rowscale,
    int M, int N, int K, int accum, int act)
{
  // Two buffers each: A tile [32][40] halfs, B tile (N-major) [64][40] halfs.
  __shared__ __align__(16) _Float16 As[2][32 * 40];
  __shared__ __align__(16) _Float16 Bs[2][64 * 40];

  const int t    = threadIdx.x;
  const int lane = t & 31;
  const int wave = t >> 5;      // 0..7
  const int wm   = wave & 1;    // M sub-tile 0..1
  const int wn   = wave >> 1;   // N sub-tile 0..3
  const int m0   = blockIdx.y * 32;
  const int n0   = blockIdx.x * 64;

  // per-thread async-copy slot for the A tile: 32 rows x 64 B = 256 lanes x 8 B
  const int am = t >> 3;          // row 0..31
  const int ak = (t & 7) * 4;     // half offset 0..28 (8-byte chunks)
  // per-thread B staging slot: row k = t>>3, 8 columns starting at (t&7)*8
  const int bk = t >> 3;
  const int bn = (t & 7) * 8;

  v8f acc = {};

  union Frag { v16h v; unsigned int u[8]; };

  auto stageA = [&](int kb, int buf) {
    unsigned lds = (unsigned)(size_t)(const void*)&As[buf][am * 40 + ak];
    unsigned long long gbl =
        (unsigned long long)(const void*)(A + (size_t)(m0 + am) * K + kb + ak);
    asm volatile("global_load_async_to_lds_b64 %0, %1, off"
                 :: "v"(lds), "v"(gbl) : "memory");
  };
  auto stageB = [&](int kb, int buf) {
    const _Float16* bp = B + (size_t)(kb + bk) * N + n0 + bn;
#pragma unroll
    for (int j = 0; j < 8; ++j)
      Bs[buf][(bn + j) * 40 + bk] = bp[j];
  };

  // ---- prologue: stage tile 0 into buffer 0 ----
  stageA(0, 0);
  stageB(0, 0);
  asm volatile("s_wait_asynccnt 0x0" ::: "memory");
  __syncthreads();

  for (int kb = 0; kb < K; kb += 32) {
    const int buf = (kb >> 5) & 1;

    // ---- stage next tile into the other buffer (overlaps with WMMA) ----
    if (kb + 32 < K) {
      stageA(kb + 32, buf ^ 1);
      stageB(kb + 32, buf ^ 1);
      if (kb + 64 < K)
        __builtin_prefetch(B + (size_t)(kb + 64) * N + n0 + bn, 0, 1);
    }

    // ---- build fragments per ISA 16-bit layout:
    // VGPR j: klo = (j>>2)*16 + ((lane>>4)&1)*8 + (j&3)*2 ; packed pair (klo,klo+1)
    Frag af, bf;
    const unsigned int* asu = (const unsigned int*)As[buf];  // 20 dwords / row
    const unsigned int* bsu = (const unsigned int*)Bs[buf];  // 20 dwords / row
    int mrow  = wm * 16 + (lane & 15);
    int ncol  = wn * 16 + (lane & 15);
    int khalf = ((lane >> 4) & 1) * 8;
#pragma unroll
    for (int j = 0; j < 8; ++j) {
      int klo = (j >> 2) * 16 + khalf + (j & 3) * 2;
      af.u[j] = asu[mrow * 20 + (klo >> 1)];
      bf.u[j] = bsu[ncol * 20 + (klo >> 1)];
    }
    acc = __builtin_amdgcn_wmma_f32_16x16x32_f16(
        false, af.v, false, bf.v, (short)0, acc, false, false);

    // next tile's async A copy + B stores must be complete & visible
    asm volatile("s_wait_asynccnt 0x0" ::: "memory");
    __syncthreads();
  }

  // ---- epilogue: C/D layout: VGPR i -> M = i + 8*(lane>=16), N = lane%16
#pragma unroll
  for (int i = 0; i < 8; ++i) {
    int row = m0 + wm * 16 + i + ((lane >> 4) << 3);
    int col = n0 + wn * 16 + (lane & 15);
    float v = acc[i];
    if (rowscale) v *= rowscale[row];
    if (bias)     v += bias[col];
    if (accum)    v += C[(size_t)row * N + col];
    if (act == 1) v = fmaxf(v, 0.f);
    else if (act == 2) v = tanhf(v);
    C[(size_t)row * N + col] = v;
  }
}

// =====================================================================
// Elementwise / utility kernels
// =====================================================================
__global__ void k_f32_to_f16(const float* __restrict__ src, _Float16* __restrict__ dst, int n) {
  int i = blockIdx.x * 256 + threadIdx.x;
  if (i < n) dst[i] = (_Float16)src[i];
}

__global__ void k_zero_f32(float* __restrict__ p, int n) {
  int i = blockIdx.x * 256 + threadIdx.x;
  if (i < n) p[i] = 0.f;
}

// Gather rows of a f32 [L,D] matrix by edge source index, converting to f16.
__global__ void k_gather_rows(const float* __restrict__ h, const int* __restrict__ idx,
                              _Float16* __restrict__ out, int n) {
  int i = blockIdx.x * 256 + threadIdx.x;
  if (i >= n) return;
  int e = i >> 10, d = i & 1023;
  out[i] = (_Float16)h[(size_t)idx[e] * D_ + d];
}

// coef[b*E+e] = a[etype[e]*NB + b]
__global__ void k_coef(const float* __restrict__ a, const int* __restrict__ etype,
                       float* __restrict__ coef) {
  int i = blockIdx.x * 256 + threadIdx.x;
  if (i >= NB_ * E_) return;
  int b = i >> 11, e = i & (E_ - 1);
  coef[i] = a[etype[e] * NB_ + b];
}

// segment-sum: agg[dst[e]] += msg[e]
__global__ void k_scatter_add(const float* __restrict__ msg, const int* __restrict__ dst,
                              float* __restrict__ agg, int n) {
  int i = blockIdx.x * 256 + threadIdx.x;
  if (i >= n) return;
  int e = i >> 10, d = i & 1023;
  atomicAdd(&agg[(size_t)dst[e] * D_ + d], msg[i]);
}

// token pooling: out[p,d] = sum_s sel[p,s]*tok[p,s,d] / sum_s sel[p,s]  (f16 out)
__global__ void k_token_reduce(const float* __restrict__ tok, const float* __restrict__ sel,
                               _Float16* __restrict__ out, int n) {
  int i = blockIdx.x * 256 + threadIdx.x;
  if (i >= n) return;
  int p = i >> 10, d = i & 1023;
  const float* tp = tok + (size_t)p * SL_ * OD_ + d;
  const float* sp = sel + (size_t)p * SL_;
  float num = 0.f, den = 0.f;
#pragma unroll
  for (int s = 0; s < SL_; ++s) { num += sp[s] * tp[(size_t)s * OD_]; den += sp[s]; }
  out[i] = (_Float16)(num / den);
}

// Windowed attention; one block per (l,h). Reads P = out_@Wqkv (L,3,H,DH),
// writes center-row output (L, H*DH) as f16 (operand for Wo GEMM).
__global__ __launch_bounds__(256) void k_attn(const float* __restrict__ P,
                                              _Float16* __restrict__ Oh) {
  int l = blockIdx.x, h = blockIdx.y, t = threadIdx.x;
  __shared__ float sred[256];
  __shared__ float sc[WW_];
  const float* q = P + ((size_t)l * 3 + 0) * (H_ * DH_) + h * DH_;
  for (int j = 0; j < WW_; ++j) {
    int src = l - SW_ + j;
    src = src < 0 ? 0 : (src > L_ - 1 ? L_ - 1 : src);
    const float* kv = P + ((size_t)src * 3 + 1) * (H_ * DH_) + h * DH_;
    float p = 0.f;
    for (int d = t; d < DH_; d += 256) p += q[d] * kv[d];
    sred[t] = p; __syncthreads();
    for (int s = 128; s > 0; s >>= 1) { if (t < s) sred[t] += sred[t + s]; __syncthreads(); }
    if (t == 0) sc[j] = sred[0] * 0.04419417382415922f;  // 1/sqrt(512)
    __syncthreads();
  }
  float mx = sc[0];
#pragma unroll
  for (int j = 1; j < WW_; ++j) mx = fmaxf(mx, sc[j]);
  float e[WW_], den = 0.f;
#pragma unroll
  for (int j = 0; j < WW_; ++j) { e[j] = expf(sc[j] - mx); den += e[j]; }
  float inv = 1.f / den;
  for (int d = t; d < DH_; d += 256) {
    float o = 0.f;
#pragma unroll
    for (int j = 0; j < WW_; ++j) {
      int src = l - SW_ + j;
      src = src < 0 ? 0 : (src > L_ - 1 ? L_ - 1 : src);
      o += e[j] * inv * P[((size_t)src * 3 + 2) * (H_ * DH_) + h * DH_ + d];
    }
    Oh[(size_t)l * (H_ * DH_) + h * DH_ + d] = (_Float16)o;
  }
}

// per-row L2 norm of relatt
__global__ __launch_bounds__(256) void k_rnorm(const float* __restrict__ x, float* __restrict__ nrm) {
  int l = blockIdx.x, t = threadIdx.x;
  __shared__ float sred[256];
  float a = 0.f;
  for (int d = t; d < D_; d += 256) { float v = x[(size_t)l * D_ + d]; a += v * v; }
  sred[t] = a; __syncthreads();
  for (int s = 128; s > 0; s >>= 1) { if (t < s) sred[t] += sred[t + s]; __syncthreads(); }
  if (t == 0) nrm[l] = sqrtf(sred[0]);
}

// cosine per (l, s*ND+d) pair: block per p in [0, L*NS*ND)
__global__ __launch_bounds__(256) void k_cos(const float* __restrict__ dst_emb,
                                             const float* __restrict__ relatt,
                                             const float* __restrict__ rnorm,
                                             float* __restrict__ cosine) {
  int p = blockIdx.x, t = threadIdx.x;
  int l = p / (NS_ * ND_);
  const float* de = dst_emb + (size_t)p * D_;
  const float* re = relatt + (size_t)l * D_;
  __shared__ float s1[256], s2[256];
  float a = 0.f, b = 0.f;
  for (int d = t; d < D_; d += 256) { float x = de[d]; a += re[d] * x; b += x * x; }
  s1[t] = a; s2[t] = b; __syncthreads();
  for (int s = 128; s > 0; s >>= 1) {
    if (t < s) { s1[t] += s1[t + s]; s2[t] += s2[t + s]; }
    __syncthreads();
  }
  if (t == 0) {
    float den = fmaxf(rnorm[l] * sqrtf(s2[0]), 1e-8f);
    cosine[p] = fabsf(s1[0] / den);
  }
}

// alpha over ND per (l,s): one thread each
__global__ void k_alpha(const float* __restrict__ cosine, const float* __restrict__ weights,
                        const float* __restrict__ sentics, const unsigned char* __restrict__ masks,
                        const float* __restrict__ src_masks, float* __restrict__ alpha) {
  int i = blockIdx.x * 256 + threadIdx.x;
  if (i >= L_ * NS_) return;
  float sm = src_masks[i];
  float om[ND_], mx = NEG_INF;
#pragma unroll
  for (int d = 0; d < ND_; ++d) {
    int p = i * ND_ + d;
    float o = masks[p] ? (0.5f * (weights[p] * cosine[p]) + 0.5f * fabsf(sentics[p])) : NEG_INF;
    om[d] = o; mx = fmaxf(mx, o);
  }
  float den = 0.f, e[ND_];
#pragma unroll
  for (int d = 0; d < ND_; ++d) { e[d] = expf(om[d] - mx); den += e[d]; }
#pragma unroll
  for (int d = 0; d < ND_; ++d) alpha[i * ND_ + d] = sm * e[d] / den;
}

// re_dot[p,d] = r_param[rels[p]][d] * alpha[p] * dst_emb[p,d]
__global__ void k_redot(const float* __restrict__ r_param, const int* __restrict__ rels,
                        const float* __restrict__ alpha, const float* __restrict__ dst_emb,
                        float* __restrict__ re_dot, int n) {
  int i = blockIdx.x * 256 + threadIdx.x;
  if (i >= n) return;
  int p = i >> 10, d = i & 1023;
  re_dot[i] = r_param[(size_t)rels[p] * D_ + d] * alpha[p] * dst_emb[i];
}

// s_score[p] = dot(src_emb[p/ND], re_dot[p]) ; block per p
__global__ __launch_bounds__(256) void k_sscore(const float* __restrict__ src_emb,
                                                const float* __restrict__ re_dot,
                                                float* __restrict__ s_score) {
  int p = blockIdx.x, t = threadIdx.x;
  const float* se = src_emb + (size_t)(p / ND_) * D_;
  const float* rd = re_dot + (size_t)p * D_;
  __shared__ float sred[256];
  float a = 0.f;
  for (int d = t; d < D_; d += 256) a += se[d] * rd[d];
  sred[t] = a; __syncthreads();
  for (int s = 128; s > 0; s >>= 1) { if (t < s) sred[t] += sred[t + s]; __syncthreads(); }
  if (t == 0) s_score[p] = sred[0];
}

__global__ void k_alpha2(const float* __restrict__ s_score, const unsigned char* __restrict__ masks,
                         const float* __restrict__ src_masks, float* __restrict__ alpha2) {
  int i = blockIdx.x * 256 + threadIdx.x;
  if (i >= L_ * NS_) return;
  float sm = src_masks[i];
  float om[ND_], mx = NEG_INF;
#pragma unroll
  for (int d = 0; d < ND_; ++d) {
    int p = i * ND_ + d;
    float o = masks[p] ? s_score[p] : NEG_INF;
    om[d] = o; mx = fmaxf(mx, o);
  }
  float den = 0.f, e[ND_];
#pragma unroll
  for (int d = 0; d < ND_; ++d) { e[d] = expf(om[d] - mx); den += e[d]; }
#pragma unroll
  for (int d = 0; d < ND_; ++d) alpha2[i * ND_ + d] = sm * e[d] / den;
}

// src_emb2[ls] = src_emb[ls] + sum_d alpha2[ls,d]*re_dot[ls,d,:]
__global__ __launch_bounds__(256) void k_srcemb2(const float* __restrict__ src_emb,
                                                 const float* __restrict__ alpha2,
                                                 const float* __restrict__ re_dot,
                                                 float* __restrict__ src_emb2) {
  int ls = blockIdx.x, t = threadIdx.x;
  for (int d = t; d < D_; d += 256) {
    float v = src_emb[(size_t)ls * D_ + d];
#pragma unroll
    for (int j = 0; j < ND_; ++j)
      v += alpha2[ls * ND_ + j] * re_dot[((size_t)ls * ND_ + j) * D_ + d];
    src_emb2[(size_t)ls * D_ + d] = v;
  }
}

__global__ __launch_bounds__(256) void k_dotsum(const float* __restrict__ src_emb2,
                                                const float* __restrict__ relatt,
                                                float* __restrict__ dot_sum) {
  int ls = blockIdx.x, t = threadIdx.x;
  const float* a = src_emb2 + (size_t)ls * D_;
  const float* b = relatt + (size_t)(ls / NS_) * D_;
  __shared__ float sred[256];
  float p = 0.f;
  for (int d = t; d < D_; d += 256) p += a[d] * b[d];
  sred[t] = p; __syncthreads();
  for (int s = 128; s > 0; s >>= 1) { if (t < s) sred[t] += sred[t + s]; __syncthreads(); }
  if (t == 0) dot_sum[ls] = sred[0];
}

__global__ __launch_bounds__(256) void k_symbolic(const unsigned char* __restrict__ masks,
                                                  const float* __restrict__ dot_sum,
                                                  const float* __restrict__ src_masks,
                                                  const float* __restrict__ src_emb2,
                                                  float* __restrict__ symbolic) {
  int l = blockIdx.x, t = threadIdx.x;
  __shared__ float att[NS_];
  if (t == 0) {
    float sc[NS_], mx = NEG_INF;
    for (int s = 0; s < NS_; ++s) {
      int cnt = 0;
      for (int d = 0; d < ND_; ++d) cnt += masks[(l * NS_ + s) * ND_ + d] ? 1 : 0;
      sc[s] = (cnt > 0) ? dot_sum[l * NS_ + s] : NEG_INF;
      mx = fmaxf(mx, sc[s]);
    }
    float den = 0.f;
    for (int s = 0; s < NS_; ++s) den += expf(sc[s] - mx);
    for (int s = 0; s < NS_; ++s)
      att[s] = (expf(sc[s] - mx) / den) * src_masks[l * NS_ + s];
  }
  __syncthreads();
  for (int d = t; d < D_; d += 256) {
    float a = 0.f;
#pragma unroll
    for (int s = 0; s < NS_; ++s)
      a += att[s] * src_emb2[((size_t)(l * NS_ + s)) * D_ + d];
    symbolic[(size_t)l * D_ + d] = a;
  }
}

// feat_h[l, 0:4096] = [out_, hidden_rgcn, relatt, symbolic] as f16
__global__ void k_feat(const float* __restrict__ o0, const float* __restrict__ o1,
                       const float* __restrict__ o2, const float* __restrict__ o3,
                       _Float16* __restrict__ feat, int n) {
  int i = blockIdx.x * 256 + threadIdx.x;
  if (i >= n) return;
  int l = i >> 12, d = i & 4095;
  int sel = d >> 10, dd = d & 1023;
  const float* s = (sel == 0) ? o0 : (sel == 1) ? o1 : (sel == 2) ? o2 : o3;
  feat[i] = (_Float16)s[(size_t)l * D_ + dd];
}

// logits (N=7 — plain VALU) + log_softmax + NLL loss accumulation
__global__ __launch_bounds__(256) void k_final(const float* __restrict__ fused,
                                               const float* __restrict__ linW,
                                               const float* __restrict__ linb,
                                               const int* __restrict__ label,
                                               float* __restrict__ out) {
  int l = blockIdx.x, t = threadIdx.x;
  int g = t >> 5, lane = t & 31;
  __shared__ float sred[256];
  __shared__ float logits[NC_];
  float p = 0.f;
  if (g < NC_) {
    const float* f = fused + (size_t)l * D_;
    for (int d = lane; d < D_; d += 32) p += f[d] * linW[d * NC_ + g];
  }
  sred[t] = p; __syncthreads();
  for (int s = 16; s > 0; s >>= 1) { if (lane < s) sred[t] += sred[t + s]; __syncthreads(); }
  if (g < NC_ && lane == 0) logits[g] = sred[t] + linb[g];
  __syncthreads();
  if (t == 0) {
    float mx = logits[0];
    for (int j = 1; j < NC_; ++j) mx = fmaxf(mx, logits[j]);
    float den = 0.f;
    for (int j = 0; j < NC_; ++j) den += expf(logits[j] - mx);
    float lse = logf(den) + mx;
    for (int j = 0; j < NC_; ++j) out[l * NC_ + j] = logits[j] - lse;
    atomicAdd(&out[L_ * NC_], -(logits[label[l]] - lse) * (1.f / ((float)L_ * (float)L_)));
  }
}

// =====================================================================
// Host orchestration
// =====================================================================
extern "C" void kernel_launch(void* const* d_in, const int* in_sizes, int n_in,
                              void* d_out_v, int out_size, void* d_ws, size_t ws_size,
                              hipStream_t stream) {
  (void)in_sizes; (void)n_in; (void)out_size; (void)ws_size;

  const float* cls_out      = (const float*)d_in[0];
  const float* src_tok_emb  = (const float*)d_in[1];
  const float* dst_tok_emb  = (const float*)d_in[2];
  const float* weights      = (const float*)d_in[3];
  const float* sentics      = (const float*)d_in[4];
  const float* src_masks    = (const float*)d_in[5];
  const float* sel_mask_src = (const float*)d_in[6];
  const float* sel_mask_dst = (const float*)d_in[7];
  const float* fw_W   = (const float*)d_in[8];
  const float* fw_b   = (const float*)d_in[9];
  const float* fwc_W  = (const float*)d_in[10];
  const float* fwc_b  = (const float*)d_in[11];
  const float* V1     = (const float*)d_in[12];
  const float* a1     = (const float*)d_in[13];
  const float* Ws1    = (const float*)d_in[14];
  const float* b1     = (const float*)d_in[15];
  const float* V2     = (const float*)d_in[16];
  const float* a2     = (const float*)d_in[17];
  const float* Ws2    = (const float*)d_in[18];
  const float* b2     = (const float*)d_in[19];
  const float* Wqkv   = (const float*)d_in[20];
  const float* bqkv   = (const float*)d_in[21];
  const float* Wo     = (const float*)d_in[22];
  const float* bo     = (const float*)d_in[23];
  const float* r_param   = (const float*)d_in[24];
  const float* fusion_W  = (const float*)d_in[25];
  const float* fusion_b  = (const float*)d_in[26];
  const float* lin_W  = (const float*)d_in[27];
  const float* lin_b  = (const float*)d_in[28];
  const int* str_src   = (const int*)d_in[29];
  const int* str_dst   = (const int*)d_in[30];
  const int* str_etype = (const int*)d_in[31];
  const int* rels      = (const int*)d_in[32];
  const unsigned char* masks = (const unsigned char*)d_in[33];
  const int* label     = (const int*)d_in[34];

  float* d_out = (float*)d_out_v;

  // ---- linear workspace allocator ----
  char* ws = (char*)d_ws;
  size_t cur = 0;
  auto alloc = [&](size_t bytes) -> void* {
    void* r = ws + cur;
    cur += (bytes + 255) & ~(size_t)255;
    return r;
  };

  _Float16* h_cls   = (_Float16*)alloc((size_t)L_ * OD_ * 2);
  _Float16* h_fwW   = (_Float16*)alloc((size_t)OD_ * D_ * 2);
  _Float16* h_fwcW  = (_Float16*)alloc((size_t)OD_ * D_ * 2);
  _Float16* h_Ws    = (_Float16*)alloc((size_t)D_ * D_ * 2);
  _Float16* h_V     = (_Float16*)alloc((size_t)NB_ * D_ * D_ * 2);
  _Float16* h_Wqkv  = (_Float16*)alloc((size_t)D_ * 3 * H_ * DH_ * 2);
  _Float16* h_Wo    = (_Float16*)alloc((size_t)H_ * DH_ * D_ * 2);
  _Float16* h_fusW  = (_Float16*)alloc((size_t)4 * D_ * D_ * 2);
  float*    out_f   = (float*)alloc((size_t)L_ * D_ * 4);
  _Float16* h_out   = (_Float16*)alloc((size_t)L_ * D_ * 2);
  float*    hid     = (float*)alloc((size_t)L_ * D_ * 4);
  _Float16* h_hid   = (_Float16*)alloc((size_t)L_ * D_ * 2);
  float*    hrg     = (float*)alloc((size_t)L_ * D_ * 4);
  _Float16* Hs_h    = (_Float16*)alloc((size_t)E_ * D_ * 2);
  float*    coef    = (float*)alloc((size_t)NB_ * E_ * 4);
  float*    msg     = (float*)alloc((size_t)E_ * D_ * 4);
  float*    P       = (float*)alloc((size_t)L_ * 3 * H_ * DH_ * 4);
  _Float16* oatt_h  = (_Float16*)alloc((size_t)L_ * H_ * DH_ * 2);
  float*    relatt  = (float*)alloc((size_t)L_ * D_ * 4);
  float*    rnorm   = (float*)alloc((size_t)L_ * 4);
  _Float16* src_red = (_Float16*)alloc((size_t)L_ * NS_ * D_ * 2);
  _Float16* dst_red = (_Float16*)alloc((size_t)L_ * NS_ * ND_ * D_ * 2);
  float*    src_emb = (float*)alloc((size_t)L_ * NS_ * D_ * 4);
  float*    dst_emb = (float*)alloc((size_t)L_ * NS_ * ND_ * D_ * 4);
  float*    cosine  = (float*)alloc((size_t)L_ * NS_ * ND_ * 4);
  float*    alpha   = (float*)alloc((size_t)L_ * NS_ * ND_ * 4);
  float*    re_dot  = (float*)alloc((size_t)L_ * NS_ * ND_ * D_ * 4);
  float*    s_score = (float*)alloc((size_t)L_ * NS_ * ND_ * 4);
  float*    alpha2  = (float*)alloc((size_t)L_ * NS_ * ND_ * 4);
  float*    semb2   = (float*)alloc((size_t)L_ * NS_ * D_ * 4);
  float*    dot_sum = (float*)alloc((size_t)L_ * NS_ * 4);
  float*    symb    = (float*)alloc((size_t)L_ * D_ * 4);
  _Float16* feat_h  = (_Float16*)alloc((size_t)L_ * 4 * D_ * 2);
  float*    fused   = (float*)alloc((size_t)L_ * D_ * 4);

  auto conv = [&](const float* s, _Float16* d, int n) {
    k_f32_to_f16<<<(n + 255) / 256, 256, 0, stream>>>(s, d, n);
  };
  auto zero = [&](float* p, int n) {
    k_zero_f32<<<(n + 255) / 256, 256, 0, stream>>>(p, n);
  };
  auto gemm = [&](const _Float16* A, const _Float16* B, float* C, const float* bias,
                  const float* rs, int M, int N, int K, int accum, int act) {
    dim3 g(N / 64, M / 32);
    k_gemm<<<g, 256, 0, stream>>>(A, B, C, bias, rs, M, N, K, accum, act);
  };

  // ---- 1. out_ = cls_out @ fw_W + fw_b ----
  conv(cls_out, h_cls, L_ * OD_);
  conv(fw_W, h_fwW, OD_ * D_);
  gemm(h_cls, h_fwW, out_f, fw_b, nullptr, L_, D_, OD_, 0, 0);
  conv(out_f, h_out, L_ * D_);

  // ---- 2. RGCN layer 1 ----
  k_gather_rows<<<(E_ * D_ + 255) / 256, 256, 0, stream>>>(out_f, str_src, Hs_h, E_ * D_);
  conv(V1, h_V, NB_ * D_ * D_);
  k_coef<<<(NB_ * E_ + 255) / 256, 256, 0, stream>>>(a1, str_etype, coef);
  zero(msg, E_ * D_);
  for (int b = 0; b < NB_; ++b)
    gemm(Hs_h, h_V + (size_t)b * D_ * D_, msg, nullptr, coef + b * E_, E_, D_, D_, 1, 0);
  zero(hid, L_ * D_);
  k_scatter_add<<<(E_ * D_ + 255) / 256, 256, 0, stream>>>(msg, str_dst, hid, E_ * D_);
  conv(Ws1, h_Ws, D_ * D_);
  gemm(h_out, h_Ws, hid, b1, nullptr, L_, D_, D_, 1, 1 /*relu*/);
  conv(hid, h_hid, L_ * D_);

  // ---- 3. RGCN layer 2 ----
  k_gather_rows<<<(E_ * D_ + 255) / 256, 256, 0, stream>>>(hid, str_src, Hs_h, E_ * D_);
  conv(V2, h_V, NB_ * D_ * D_);
  k_coef<<<(NB_ * E_ + 255) / 256, 256, 0, stream>>>(a2, str_etype, coef);
  zero(msg, E_ * D_);
  for (int b = 0; b < NB_; ++b)
    gemm(Hs_h, h_V + (size_t)b * D_ * D_, msg, nullptr, coef + b * E_, E_, D_, D_, 1, 0);
  zero(hrg, L_ * D_);
  k_scatter_add<<<(E_ * D_ + 255) / 256, 256, 0, stream>>>(msg, str_dst, hrg, E_ * D_);
  conv(Ws2, h_Ws, D_ * D_);
  gemm(h_hid, h_Ws, hrg, b2, nullptr, L_, D_, D_, 1, 0);

  // ---- 4. windowed attention (qkv shared across windows) ----
  conv(Wqkv, h_Wqkv, D_ * 3 * H_ * DH_);
  gemm(h_out, h_Wqkv, P, bqkv, nullptr, L_, 3 * H_ * DH_, D_, 0, 0);
  k_attn<<<dim3(L_, H_), 256, 0, stream>>>(P, oatt_h);
  conv(Wo, h_Wo, H_ * DH_ * D_);
  gemm(oatt_h, h_Wo, relatt, bo, nullptr, L_, D_, H_ * DH_, 0, 0);

  // ---- 5. concept pooling + projection ----
  conv(fwc_W, h_fwcW, OD_ * D_);
  k_token_reduce<<<(L_ * NS_ * D_ + 255) / 256, 256, 0, stream>>>(
      src_tok_emb, sel_mask_src, src_red, L_ * NS_ * D_);
  gemm(src_red, h_fwcW, src_emb, fwc_b, nullptr, L_ * NS_, D_, OD_, 0, 0);
  k_token_reduce<<<(L_ * NS_ * ND_ * D_ + 255) / 256, 256, 0, stream>>>(
      dst_tok_emb, sel_mask_dst, dst_red, L_ * NS_ * ND_ * D_);
  gemm(dst_red, h_fwcW, dst_emb, fwc_b, nullptr, L_ * NS_ * ND_, D_, OD_, 0, 0);

  // ---- 6. symbolic attention chain ----
  k_rnorm<<<L_, 256, 0, stream>>>(relatt, rnorm);
  k_cos<<<L_ * NS_ * ND_, 256, 0, stream>>>(dst_emb, relatt, rnorm, cosine);
  k_alpha<<<(L_ * NS_ + 255) / 256, 256, 0, stream>>>(cosine, weights, sentics, masks, src_masks, alpha);
  k_redot<<<(L_ * NS_ * ND_ * D_ + 255) / 256, 256, 0, stream>>>(
      r_param, rels, alpha, dst_emb, re_dot, L_ * NS_ * ND_ * D_);
  k_sscore<<<L_ * NS_ * ND_, 256, 0, stream>>>(src_emb, re_dot, s_score);
  k_alpha2<<<(L_ * NS_ + 255) / 256, 256, 0, stream>>>(s_score, masks, src_masks, alpha2);
  k_srcemb2<<<L_ * NS_, 256, 0, stream>>>(src_emb, alpha2, re_dot, semb2);
  k_dotsum<<<L_ * NS_, 256, 0, stream>>>(semb2, relatt, dot_sum);
  k_symbolic<<<L_, 256, 0, stream>>>(masks, dot_sum, src_masks, semb2, symb);

  // ---- 7. fusion + classifier + loss ----
  k_feat<<<(L_ * 4 * D_ + 255) / 256, 256, 0, stream>>>(out_f, hrg, relatt, symb, feat_h, L_ * 4 * D_);
  conv(fusion_W, h_fusW, 4 * D_ * D_);
  gemm(feat_h, h_fusW, fused, fusion_b, nullptr, L_, D_, 4 * D_, 0, 2 /*tanh*/);
  zero(d_out + L_ * NC_, 1);
  k_final<<<L_, 256, 0, stream>>>(fused, lin_W, lin_b, label, d_out);
}
